// MultiHeadAttention_21285857919382
// MI455X (gfx1250) — compile-verified, matrix operand order checked by reading, not run on hardware
//
#include <hip/hip_runtime.h>
#include <hip/hip_bf16.h>
#include <stdint.h>

#define D_MODEL 1024
#define NHEADS  16
#define DK      64
#define SEQ     2048
#define BATCH   4

typedef __attribute__((ext_vector_type(16))) __bf16 bf16x16;
typedef __attribute__((ext_vector_type(8)))  __bf16 bf16x8;
typedef __attribute__((ext_vector_type(8)))  float  f32x8;

static __device__ __forceinline__ unsigned short f2bf(float f) {
  union { float f; unsigned int u; } v; v.f = f;
  unsigned int u = v.u;
  return (unsigned short)((u + 0x7FFFu + ((u >> 16) & 1u)) >> 16);
}

// Build a 16x32 bf16 WMMA operand fragment from LDS.
// Lane (l&15) selects row/col; kb=(l>>4)*8 selects the K-phase.
// Elements 0..7  = K { base+0 .. base+7 }
// Elements 8..15 = K { base+16 .. base+23 }   (per CDNA5 16-bit A/B layout)
static __device__ __forceinline__ bf16x16 ld_frag(const unsigned short* p) {
  bf16x8 lo = *(const bf16x8*)(p);
  bf16x8 hi = *(const bf16x8*)(p + 16);
  return __builtin_shufflevector(lo, hi, 0,1,2,3,4,5,6,7,8,9,10,11,12,13,14,15);
}

__global__ void __launch_bounds__(256)
cvt_f32_bf16(const float* __restrict__ x, unsigned short* __restrict__ y, int n) {
  int i = blockIdx.x * blockDim.x + threadIdx.x;
  if (i < n) y[i] = f2bf(x[i]);
}

// 1024x1024 f32 -> bf16 with transpose (out[n][k] = in[k][n]), LDS-tiled.
__global__ void __launch_bounds__(256)
cvt_transpose_bf16(const float* __restrict__ in, unsigned short* __restrict__ out) {
  __shared__ unsigned short t[32][33];
  const int bx = blockIdx.x * 32;        // input col
  const int by = blockIdx.y * 32;        // input row
  const int x  = threadIdx.x & 31;
  const int y  = threadIdx.x >> 5;       // 0..7
  #pragma unroll
  for (int i = 0; i < 4; i++) {
    int r = y + i * 8;
    t[r][x] = f2bf(in[(size_t)(by + r) * D_MODEL + bx + x]);
  }
  __syncthreads();
  #pragma unroll
  for (int i = 0; i < 4; i++) {
    int r = y + i * 8;
    out[(size_t)(bx + r) * D_MODEL + by + x] = t[x][r];
  }
}

// C = A(bf16, [SEQ x 1024] per batch) x W^T(bf16, [N=1024 x K=1024], pre-transposed)
// 128x128 block tile, 8 waves each computing 32x64 (2x4 WMMA tiles), LDS double-buffered.
// MODE 0: bf16 head-split [B][NHEADS][SEQ][DK]        (Q, K)
// MODE 1: f32 [B][SEQ][1024]                          (final output)
// MODE 2: bf16 head-split transposed [B][NHEADS][DK][SEQ]  (V, packed 16-B stores)
template <int MODE>
__global__ void __launch_bounds__(256)
gemm_bf16_1024(const unsigned short* __restrict__ A,
               const unsigned short* __restrict__ Wt,
               unsigned short* __restrict__ outBF,
               float* __restrict__ outF) {
  __shared__ unsigned short As[2][128 * 40]; // rows x K32, stride 40 (pad)
  __shared__ unsigned short Bs[2][128 * 40]; // cols x K32, stride 40

  const int tid  = threadIdx.x;
  const int lane = tid & 31;
  const int wave = tid >> 5;
  const int wm   = wave >> 1;   // 0..3 : 32-row group
  const int wn   = wave & 1;    // 0..1 : 64-col group
  const int r16  = lane & 15;
  const int kb   = (lane >> 4) * 8;
  const int hs   = kb;

  const int n0 = blockIdx.x * 128;
  const int m0 = blockIdx.y * 128;
  const int b  = blockIdx.z;

  const unsigned short* Ab = A + (size_t)b * SEQ * D_MODEL;

  f32x8 acc[2][4] = {};

  const int lr = tid >> 1;          // 0..127
  const int lc = (tid & 1) * 16;    // 0 or 16

  const uint4* gA = (const uint4*)(Ab + (size_t)(m0 + lr) * D_MODEL + lc);
  const uint4* gB = (const uint4*)(Wt + (size_t)(n0 + lr) * D_MODEL + lc);

  uint4 a0 = gA[0], a1 = gA[1];
  uint4 b0 = gB[0], b1 = gB[1];
  *(uint4*)&As[0][lr * 40 + lc]     = a0;
  *(uint4*)&As[0][lr * 40 + lc + 8] = a1;
  *(uint4*)&Bs[0][lr * 40 + lc]     = b0;
  *(uint4*)&Bs[0][lr * 40 + lc + 8] = b1;

  const int NIT = D_MODEL / 32;
  for (int it = 0; it < NIT; ++it) {
    __syncthreads();
    const int cur = it & 1;
    const bool more = (it + 1 < NIT);
    if (more) {                       // issue next-tile global loads early
      const uint4* pa = gA + (size_t)(it + 1) * 4;
      const uint4* pb = gB + (size_t)(it + 1) * 4;
      a0 = pa[0]; a1 = pa[1];
      b0 = pb[0]; b1 = pb[1];
    }

    bf16x16 af[2], bf[4];
    #pragma unroll
    for (int mi = 0; mi < 2; mi++)
      af[mi] = ld_frag(&As[cur][(wm * 32 + mi * 16 + r16) * 40 + kb]);
    #pragma unroll
    for (int ni = 0; ni < 4; ni++)
      bf[ni] = ld_frag(&Bs[cur][(wn * 64 + ni * 16 + r16) * 40 + kb]);
    #pragma unroll
    for (int mi = 0; mi < 2; mi++)
      #pragma unroll
      for (int ni = 0; ni < 4; ni++)
        acc[mi][ni] = __builtin_amdgcn_wmma_f32_16x16x32_bf16(
            false, af[mi], false, bf[ni], (short)0, acc[mi][ni], false, false);

    if (more) {
      const int nxt = cur ^ 1;
      *(uint4*)&As[nxt][lr * 40 + lc]     = a0;
      *(uint4*)&As[nxt][lr * 40 + lc + 8] = a1;
      *(uint4*)&Bs[nxt][lr * 40 + lc]     = b0;
      *(uint4*)&Bs[nxt][lr * 40 + lc + 8] = b1;
    }
  }

  #pragma unroll
  for (int mi = 0; mi < 2; mi++) {
    #pragma unroll
    for (int ni = 0; ni < 4; ni++) {
      const int colg = n0 + wn * 64 + ni * 16 + r16;
      const int rowb = m0 + wm * 32 + mi * 16 + hs;
      if constexpr (MODE == 0) {
        const int h = colg >> 6, c = colg & 63;
        #pragma unroll
        for (int r = 0; r < 8; r++)
          outBF[(((size_t)b * NHEADS + h) * SEQ + rowb + r) * DK + c] =
              f2bf(acc[mi][ni][r]);
      } else if constexpr (MODE == 1) {
        #pragma unroll
        for (int r = 0; r < 8; r++)
          outF[((size_t)b * SEQ + rowb + r) * D_MODEL + colg] = acc[mi][ni][r];
      } else { // MODE 2: [B][NHEADS][DK][SEQ], 8 consecutive rows -> one 16-B store
        const int h = colg >> 6, c = colg & 63;
        union { unsigned short s[8]; uint4 v; } pk;
        #pragma unroll
        for (int r = 0; r < 8; r++) pk.s[r] = f2bf(acc[mi][ni][r]);
        *(uint4*)&outBF[(((size_t)b * NHEADS + h) * DK + c) * SEQ + rowb] = pk.v;
      }
    }
  }
}

// Fused causal flash attention per (batch, head, 64-query tile).
// Q/K: bf16 [B][NHEADS][SEQ][DK]; Vt: bf16 [B][NHEADS][DK][SEQ] (pre-transposed).
// AO: bf16 merged [B][SEQ][D_MODEL].
__global__ void __launch_bounds__(256)
flash_attn_bf16(const unsigned short* __restrict__ Q,
                const unsigned short* __restrict__ K,
                const unsigned short* __restrict__ Vt,
                unsigned short* __restrict__ AO) {
  __shared__ unsigned short Qs [64 * 72];
  __shared__ unsigned short Ks [64 * 72];
  __shared__ unsigned short Vts[64 * 72]; // [dk][key]
  __shared__ unsigned short Ps [64 * 72];
  __shared__ float Ssc[64 * 66];
  __shared__ float mrow[64], lrow[64], fac[64], linv[64];

  const int tid  = threadIdx.x;
  const int lane = tid & 31;
  const int wave = tid >> 5;
  const int wm   = wave >> 1;     // 16-row tile of S/O
  const int wn   = wave & 1;      // 32-col half
  const int r16  = lane & 15;
  const int kb   = (lane >> 4) * 8;
  const int hs   = kb;

  const int qt = blockIdx.x;
  const int h  = blockIdx.y;
  const int b  = blockIdx.z;

  const size_t headBase = ((size_t)b * NHEADS + h) * SEQ * DK;
  const unsigned short* Qh = Q + headBase + (size_t)qt * 64 * DK;
  const unsigned short* Kh = K + headBase;
  const unsigned short* Vh = Vt + headBase; // [DK][SEQ]

  { // load Q tile
    int r = tid >> 2, c = (tid & 3) * 16;
    const uint4* g = (const uint4*)(Qh + r * DK + c);
    uint4 q0 = g[0], q1 = g[1];
    *(uint4*)&Qs[r * 72 + c]     = q0;
    *(uint4*)&Qs[r * 72 + c + 8] = q1;
  }
  if (tid < 64) { mrow[tid] = -3.0e38f; lrow[tid] = 0.0f; }

  f32x8 oacc[2] = {};

  const int myrow = tid >> 2;
  const int q4    = tid & 3;
  const int c0    = q4 * 16;
  const int grow  = qt * 64 + myrow;

  for (int j = 0; j <= qt; j++) {
    { // load K tile (row-major) and V tile (already transposed in memory)
      int r = tid >> 2, c = (tid & 3) * 16;
      const uint4* gk = (const uint4*)(Kh + (size_t)(j * 64 + r) * DK + c);
      uint4 k0v = gk[0], k1v = gk[1];
      *(uint4*)&Ks[r * 72 + c]     = k0v;
      *(uint4*)&Ks[r * 72 + c + 8] = k1v;

      const uint4* gv = (const uint4*)(Vh + (size_t)r * SEQ + j * 64 + c);
      uint4 v0 = gv[0], v1 = gv[1];
      *(uint4*)&Vts[r * 72 + c]     = v0;   // row r = dk, cols = keys
      *(uint4*)&Vts[r * 72 + c + 8] = v1;
    }
    __syncthreads();

    // S = Q * K^T  (Ks row-major == B^T layout for WMMA)
    f32x8 sacc[2] = {};
    #pragma unroll
    for (int k0 = 0; k0 < DK; k0 += 32) {
      bf16x16 a = ld_frag(&Qs[(wm * 16 + r16) * 72 + k0 + kb]);
      #pragma unroll
      for (int i = 0; i < 2; i++) {
        int ct = wn * 2 + i;
        bf16x16 bb = ld_frag(&Ks[(ct * 16 + r16) * 72 + k0 + kb]);
        sacc[i] = __builtin_amdgcn_wmma_f32_16x16x32_bf16(
            false, a, false, bb, (short)0, sacc[i], false, false);
      }
    }
    #pragma unroll
    for (int i = 0; i < 2; i++) {
      int ct = wn * 2 + i;
      #pragma unroll
      for (int r = 0; r < 8; r++)
        Ssc[(wm * 16 + hs + r) * 66 + ct * 16 + r16] = sacc[i][r];
    }
    __syncthreads();

    { // online softmax: 4 lanes per query row, 16 cols each
      float sv[16];
      float mx = -3.0e38f;
      #pragma unroll
      for (int c = 0; c < 16; c++) {
        float s = Ssc[myrow * 66 + c0 + c] * 0.125f;           // 1/sqrt(64)
        if (j * 64 + c0 + c > grow) s = -3.0e38f;              // causal mask
        sv[c] = s;
        mx = fmaxf(mx, s);
      }
      mx = fmaxf(mx, __shfl_xor(mx, 1));
      mx = fmaxf(mx, __shfl_xor(mx, 2));
      float mold = mrow[myrow];
      float mnew = fmaxf(mold, mx);
      float f = __expf(mold - mnew);
      float sum = 0.0f;
      #pragma unroll
      for (int c = 0; c < 16; c++) {
        float p = __expf(sv[c] - mnew);
        Ps[myrow * 72 + c0 + c] = f2bf(p);
        sum += p;
      }
      sum += __shfl_xor(sum, 1);
      sum += __shfl_xor(sum, 2);
      if (q4 == 0) {
        mrow[myrow] = mnew;
        lrow[myrow] = lrow[myrow] * f + sum;
        fac[myrow]  = f;
      }
    }
    __syncthreads();

    // rescale O, then O += P * V
    #pragma unroll
    for (int i = 0; i < 2; i++)
      #pragma unroll
      for (int r = 0; r < 8; r++)
        oacc[i][r] *= fac[wm * 16 + hs + r];

    #pragma unroll
    for (int k0 = 0; k0 < 64; k0 += 32) {
      bf16x16 a = ld_frag(&Ps[(wm * 16 + r16) * 72 + k0 + kb]);
      #pragma unroll
      for (int i = 0; i < 2; i++) {
        int ct = wn * 2 + i;
        bf16x16 bb = ld_frag(&Vts[(ct * 16 + r16) * 72 + k0 + kb]);
        oacc[i] = __builtin_amdgcn_wmma_f32_16x16x32_bf16(
            false, a, false, bb, (short)0, oacc[i], false, false);
      }
    }
    __syncthreads();
  }

  if (tid < 64) linv[tid] = 1.0f / lrow[tid];
  __syncthreads();

  #pragma unroll
  for (int i = 0; i < 2; i++) {
    int col = (wn * 2 + i) * 16 + r16;
    #pragma unroll
    for (int r = 0; r < 8; r++) {
      int row = wm * 16 + hs + r;
      float v = oacc[i][r] * linv[row];
      AO[((size_t)b * SEQ + qt * 64 + row) * D_MODEL + h * DK + col] = f2bf(v);
    }
  }
}

extern "C" void kernel_launch(void* const* d_in, const int* in_sizes, int n_in,
                              void* d_out, int out_size, void* d_ws, size_t ws_size,
                              hipStream_t stream) {
  const float* H  = (const float*)d_in[0];
  const float* Wq = (const float*)d_in[1];
  const float* Wk = (const float*)d_in[2];
  const float* Wv = (const float*)d_in[3];
  const float* Wo = (const float*)d_in[4];
  float* out = (float*)d_out;

  const size_t nH = (size_t)BATCH * SEQ * D_MODEL; // 8388608
  const size_t nW = (size_t)D_MODEL * D_MODEL;     // 1048576

  unsigned short* p  = (unsigned short*)d_ws;
  unsigned short* Hb  = p; p += nH;
  unsigned short* Wqt = p; p += nW;
  unsigned short* Wkt = p; p += nW;
  unsigned short* Wvt = p; p += nW;
  unsigned short* Wot = p; p += nW;
  unsigned short* Qb  = p; p += nH;
  unsigned short* Kb  = p; p += nH;
  unsigned short* Vtb = p; p += nH;
  unsigned short* AOb = p; p += nH;

  cvt_f32_bf16<<<(int)(nH / 256), 256, 0, stream>>>(H, Hb, (int)nH);
  dim3 gt(32, 32);
  cvt_transpose_bf16<<<gt, 256, 0, stream>>>(Wq, Wqt);
  cvt_transpose_bf16<<<gt, 256, 0, stream>>>(Wk, Wkt);
  cvt_transpose_bf16<<<gt, 256, 0, stream>>>(Wv, Wvt);
  cvt_transpose_bf16<<<gt, 256, 0, stream>>>(Wo, Wot);

  dim3 gg(D_MODEL / 128, SEQ / 128, BATCH);
  gemm_bf16_1024<0><<<gg, 256, 0, stream>>>(Hb, Wqt, Qb,  nullptr);
  gemm_bf16_1024<0><<<gg, 256, 0, stream>>>(Hb, Wkt, Kb,  nullptr);
  gemm_bf16_1024<2><<<gg, 256, 0, stream>>>(Hb, Wvt, Vtb, nullptr);

  dim3 ga(SEQ / 64, NHEADS, BATCH);
  flash_attn_bf16<<<ga, 256, 0, stream>>>(Qb, Kb, Vtb, AOb);

  gemm_bf16_1024<1><<<gg, 256, 0, stream>>>(AOb, Wot, nullptr, out);
}
